// MultiHeadAttention_32229434589905
// MI455X (gfx1250) — matrix operand order repaired and checked by reading, not compiled, not run on hardware
//
#include <hip/hip_runtime.h>
#include <hip/hip_bf16.h>
#include <math.h>

// ---------------------------------------------------------------------------
// Fused multi-head attention for MI455X (gfx1250, wave32, WMMA, async-LDS).
//   B=1024, S=128, D=512, NH=8, DH=64
// One workgroup (256 threads = 8 waves) per batch element.
// LDS (~293 KB of the WGP's 320 KB):
//   lx  : x_seq[b] as f16, row-major [128][512]             (128 KB)
//   lw0 : weight slab ping [64][512] f16                    ( 64 KB)
//   lw1 : weight slab pong [64][512] f16                    ( 64 KB)
//   lk  : K_h f16 [128][64], lv : V_h f16 [128][64]         ( 32 KB)
//   + small f32 buffers (q, x_non row, qk, biases, reduction scratch)
// K/V projections use v_wmma_f32_16x16x32_f16; x_seq is read from HBM once.
// Weight slabs (f16, pre-converted into d_ws) stream from L2 via
// GLOBAL_LOAD_ASYNC_TO_LDS_B128, double-buffered against the WMMA phases
// with counted s_wait_asynccnt (async loads complete in order).
// ---------------------------------------------------------------------------

typedef _Float16 f16;
typedef __attribute__((ext_vector_type(8)))  _Float16 v8h;
typedef __attribute__((ext_vector_type(16))) _Float16 v16h;
typedef __attribute__((ext_vector_type(8)))  float    v8f;
typedef __attribute__((ext_vector_type(4)))  int      v4i;

constexpr int Bn = 1024;
constexpr int Sn = 128;
constexpr int Dn = 512;
constexpr int NHn = 8;
constexpr int DHn = 64;

constexpr size_t SMEM_F16 = (size_t)Sn * Dn + 2 * (size_t)DHn * Dn + 2 * (size_t)Sn * DHn;
constexpr size_t SMEM_F32 = (size_t)Dn + Dn + Sn + DHn + DHn + 64;
constexpr size_t SMEM_BYTES = SMEM_F16 * 2 + SMEM_F32 * 4;   // 300288 B < 320 KB

#if defined(__HIP_DEVICE_COMPILE__) && \
    __has_builtin(__builtin_amdgcn_global_load_async_to_lds_b128) && \
    __has_builtin(__builtin_amdgcn_s_wait_asynccnt)
#define USE_ASYNC_LDS 1
#else
#define USE_ASYNC_LDS 0
#endif

#if USE_ASYNC_LDS
#define ASYNC_WAIT(n) __builtin_amdgcn_s_wait_asynccnt(n)
#else
#define ASYNC_WAIT(n) ((void)0)
#endif

// Per-wave async ops per slab: (64*512 f16)*2B / 16B per b128 / 8 waves·32 ln = 16
// (each of the 16 loop trips issues one wave-wide async instruction)

// Issue a 64x512 f16 slab copy global->LDS (async when available).
__device__ __forceinline__ void issue_slab_f16(const f16* __restrict__ gsrc,
                                               f16* __restrict__ ldst, int tid) {
#if USE_ASYNC_LDS
  // Builtin signature (from clang diagnostic): param0 = v4i in AS1 (__device__),
  // param1 = v4i in AS3 (LDS), then imm offset, imm cpol.
  typedef __attribute__((address_space(1))) v4i g_v4i;
  typedef __attribute__((address_space(3))) v4i l_v4i;
  g_v4i* g = (g_v4i*)(uintptr_t)gsrc;              // generic==AS1 bit pattern
  l_v4i* l = (l_v4i*)(uint32_t)(uintptr_t)ldst;    // low 32 bits = LDS offset
  for (int i = tid; i < (DHn * Dn) / 8; i += 256)
    __builtin_amdgcn_global_load_async_to_lds_b128(g + i, l + i, 0, 0);
#else
  for (int i = tid; i < (DHn * Dn) / 8; i += 256)
    *(v8h*)(ldst + i * 8) = *(const v8h*)(gsrc + i * 8);
#endif
}

// Fallback: convert a 64x512 fp32 slab to f16 in LDS (synchronous).
__device__ __forceinline__ void slab_from_f32(const float* __restrict__ gsrc,
                                              f16* __restrict__ ldst, int tid) {
  const float4* w4 = (const float4*)gsrc;
  for (int i = tid; i < (DHn * Dn) / 4; i += 256) {
    float4 v = w4[i];
    f16* dst = ldst + i * 4;
    dst[0] = (f16)v.x; dst[1] = (f16)v.y; dst[2] = (f16)v.z; dst[3] = (f16)v.w;
  }
}

// X(128x512,f16 LDS) @ W^T(512x64) -> lkv(128x64,f16) with f32 accum + bias.
// Wave `wave` handles M rows [wave*16, wave*16+16); all 4 N tiles.
__device__ __forceinline__ void wmma_proj(const f16* __restrict__ lx,
                                          const f16* __restrict__ lw,
                                          f16* __restrict__ lkv,
                                          const float* __restrict__ lbias,
                                          int wave, int lane) {
  const int m0 = wave * 16;
  const int r  = lane & 15;   // row (A) / column (B,D) selector within tile
  const int hi = lane >> 4;   // half-wave selector
  const f16* arow = lx + (m0 + r) * Dn;

  v8f acc0 = {}, acc1 = {}, acc2 = {}, acc3 = {};
#pragma unroll
  for (int k0 = 0; k0 < Dn; k0 += 32) {
    // A fragment (16x32 f16): lane<16 holds K k0..k0+15,
    // lane>=16 holds K k0+16..k0+31 (contiguous, matching B)
    v16h a = *(const v16h*)(arow + k0 + hi * 16);
    // B fragment (32x16 f16): column n of B == row n of W (row-major),
    // lane<16 holds K k0..k0+15, lane>=16 holds K k0+16..k0+31 (contiguous).
    const f16* bbase = lw + r * Dn + k0 + hi * 16;
    v16h b0 = *(const v16h*)(bbase + 0 * 16 * Dn);
    v16h b1 = *(const v16h*)(bbase + 1 * 16 * Dn);
    v16h b2 = *(const v16h*)(bbase + 2 * 16 * Dn);
    v16h b3 = *(const v16h*)(bbase + 3 * 16 * Dn);
    acc0 = __builtin_amdgcn_wmma_f32_16x16x32_f16(false, a, false, b0, (short)0, acc0, false, false);
    acc1 = __builtin_amdgcn_wmma_f32_16x16x32_f16(false, a, false, b1, (short)0, acc1, false, false);
    acc2 = __builtin_amdgcn_wmma_f32_16x16x32_f16(false, a, false, b2, (short)0, acc2, false, false);
    acc3 = __builtin_amdgcn_wmma_f32_16x16x32_f16(false, a, false, b3, (short)0, acc3, false, false);
  }
  // D fragment: lane holds column n0 + r, rows m0 + rr + hi*8 (rr = VGPR idx)
#pragma unroll
  for (int rr = 0; rr < 8; ++rr) {
    const int m = m0 + rr + hi * 8;
    lkv[m * DHn +  0 + r] = (f16)(acc0[rr] + lbias[ 0 + r]);
    lkv[m * DHn + 16 + r] = (f16)(acc1[rr] + lbias[16 + r]);
    lkv[m * DHn + 32 + r] = (f16)(acc2[rr] + lbias[32 + r]);
    lkv[m * DHn + 48 + r] = (f16)(acc3[rr] + lbias[48 + r]);
  }
}

// Pre-pass: convert Wk/Wv (fp32) -> f16 in workspace for byte-copyable slabs.
__global__ __launch_bounds__(256)
void convert_weights_kernel(const float* __restrict__ Wk, const float* __restrict__ Wv,
                            f16* __restrict__ wk16, f16* __restrict__ wv16) {
  const int i = blockIdx.x * 256 + threadIdx.x;   // float4 index, Dn*Dn/4 total
  if (i < Dn * Dn / 4) {
    float4 a = ((const float4*)Wk)[i];
    float4 b = ((const float4*)Wv)[i];
    f16* dk = wk16 + i * 4;
    dk[0] = (f16)a.x; dk[1] = (f16)a.y; dk[2] = (f16)a.z; dk[3] = (f16)a.w;
    f16* dv = wv16 + i * 4;
    dv[0] = (f16)b.x; dv[1] = (f16)b.y; dv[2] = (f16)b.z; dv[3] = (f16)b.w;
  }
}

__global__ __launch_bounds__(256, 1)
void mha_fused_kernel(const float* __restrict__ x_non,
                      const float* __restrict__ x_seq,
                      const int*   __restrict__ mask,
                      const float* __restrict__ Wq, const float* __restrict__ bq,
                      const float* __restrict__ Wk, const float* __restrict__ bk,
                      const float* __restrict__ Wv, const float* __restrict__ bv,
                      const f16* __restrict__ wk16, const f16* __restrict__ wv16,
                      int use16,
                      float* __restrict__ out) {
  extern __shared__ __align__(128) char smem_raw[];
  f16*   lx    = (f16*)smem_raw;            // [128][512]
  f16*   lw0   = lx + Sn * Dn;              // [64][512] ping
  f16*   lw1   = lw0 + DHn * Dn;            // [64][512] pong
  f16*   lk    = lw1 + DHn * Dn;            // [128][64]
  f16*   lv    = lk + Sn * DHn;             // [128][64]
  float* lq    = (float*)(lv + Sn * DHn);   // [512] q vector (f32)
  float* lxn   = lq + Dn;                   // [512] x_non row
  float* lqk   = lxn + Dn;                  // [128] logits / exp weights
  float* lbk   = lqk + Sn;                  // [64]
  float* lbv   = lbk + DHn;                 // [64]
  float* lred  = lbv + DHn;                 // [64] reduction scratch

  const int b    = blockIdx.x;
  const int tid  = threadIdx.x;
  const int lane = tid & 31;
  const int wave = tid >> 5;

  // ---- Stage x_seq[b] (fp32 -> f16) into LDS: the ONLY HBM read of x_seq ---
  const float4* xs4 = (const float4*)(x_seq + (size_t)b * Sn * Dn);
  for (int i = tid; i < Sn * Dn / 4; i += 256) {
    float4 v = xs4[i];
    f16* dst = lx + i * 4;
    dst[0] = (f16)v.x; dst[1] = (f16)v.y; dst[2] = (f16)v.z; dst[3] = (f16)v.w;
  }
  for (int i = tid; i < Dn; i += 256) lxn[i] = x_non[(size_t)b * Dn + i];

  // ---- Preamble: kick off Wk_0 slab into lw0 (overlaps q GEMV below) ------
  if (use16) issue_slab_f16(wk16, lw0, tid);
  else       slab_from_f32(Wk, lw0, tid);
  __syncthreads();   // lxn visible for q GEMV

  // ---- q = x_non[b] @ Wq^T + bq (f32 GEMV; 0.5 MFLOP, L2-resident Wq) -----
  for (int j = tid; j < Dn; j += 256) {
    const float* wrow = Wq + (size_t)j * Dn;
    float acc = bq[j];
#pragma unroll 8
    for (int d = 0; d < Dn; ++d) acc += lxn[d] * wrow[d];
    lq[j] = acc;
  }

  for (int h = 0; h < NHn; ++h) {
    // Biases + issue Wv_h slab into lw1 (overlaps the K_h WMMA phase).
    if (tid < DHn) {
      lbk[tid] = bk[h * DHn + tid];
      lbv[tid] = bv[h * DHn + tid];
    }
    if (use16) issue_slab_f16(wv16 + (size_t)h * DHn * Dn, lw1, tid);
    else       slab_from_f32(Wv + (size_t)h * DHn * Dn, lw1, tid);

    ASYNC_WAIT(16);        // Wk_h (older 16 ops) landed; Wv_h may be in flight
    __syncthreads();

    // ---- K_h = X @ Wk_h^T via WMMA (lw0) ----
    wmma_proj(lx, lw0, lk, lbk, wave, lane);
    __syncthreads();       // everyone done reading lw0

    // Issue Wk_{h+1} into lw0 (overlaps the V_h WMMA + attention math).
    if (h + 1 < NHn) {
      if (use16) issue_slab_f16(wk16 + (size_t)(h + 1) * DHn * Dn, lw0, tid);
      else       slab_from_f32(Wk + (size_t)(h + 1) * DHn * Dn, lw0, tid);
      ASYNC_WAIT(16);      // Wv_h (older) landed; Wk_{h+1} may be in flight
    } else {
      ASYNC_WAIT(0);       // last head: just make sure Wv_7 landed
    }
    __syncthreads();

    // ---- V_h = X @ Wv_h^T via WMMA (lw1) ----
    wmma_proj(lx, lw1, lv, lbv, wave, lane);
    __syncthreads();

    // ---- logits: qk[s] = floor((q_h . K_h[s]) / 8), mask -> -1e9 ----
    const float* qh = lq + h * DHn;
    if (tid < Sn) {
      const f16* krow = lk + tid * DHn;
      float acc = 0.f;
#pragma unroll 16
      for (int d = 0; d < DHn; ++d) acc += qh[d] * (float)krow[d];
      acc = floorf(acc * 0.125f);  // 1/sqrt(DH) = 1/8
      if (mask[((size_t)b * NHn + h) * Sn + tid] == 0) acc = -1e9f;
      lqk[tid] = acc;
    }
    __syncthreads();

    // ---- softmax over S=128 (tree reductions in LDS) ----
    if (tid < 64) lred[tid] = fmaxf(lqk[tid], lqk[tid + 64]);
    __syncthreads();
    if (tid < 32) lred[tid] = fmaxf(lred[tid], lred[tid + 32]);
    __syncthreads();
    if (tid < 16) lred[tid] = fmaxf(lred[tid], lred[tid + 16]);
    __syncthreads();
    if (tid < 8)  lred[tid] = fmaxf(lred[tid], lred[tid + 8]);
    __syncthreads();
    if (tid < 4)  lred[tid] = fmaxf(lred[tid], lred[tid + 4]);
    __syncthreads();
    if (tid < 2)  lred[tid] = fmaxf(lred[tid], lred[tid + 2]);
    __syncthreads();
    const float rowmax = fmaxf(lred[0], lred[1]);
    if (tid < Sn) lqk[tid] = __expf(lqk[tid] - rowmax);
    __syncthreads();
    if (tid < 64) lred[tid] = lqk[tid] + lqk[tid + 64];
    __syncthreads();
    if (tid < 32) lred[tid] += lred[tid + 32];
    __syncthreads();
    if (tid < 16) lred[tid] += lred[tid + 16];
    __syncthreads();
    if (tid < 8)  lred[tid] += lred[tid + 8];
    __syncthreads();
    if (tid < 4)  lred[tid] += lred[tid + 4];
    __syncthreads();
    if (tid < 2)  lred[tid] += lred[tid + 2];
    __syncthreads();
    const float inv = 1.0f / (lred[0] + lred[1]);

    // ---- z_h[d] = (attn . V_h[:,d]) ----
    if (tid < DHn) {
      float acc = 0.f;
#pragma unroll 8
      for (int s = 0; s < Sn; ++s) acc += lqk[s] * (float)lv[s * DHn + tid];
      out[(size_t)b * Dn + h * DHn + tid] = acc * inv;
    }
    __syncthreads();
  }
}

extern "C" void kernel_launch(void* const* d_in, const int* in_sizes, int n_in,
                              void* d_out, int out_size, void* d_ws, size_t ws_size,
                              hipStream_t stream) {
  const float* x_non = (const float*)d_in[0];
  const float* x_seq = (const float*)d_in[1];
  const int*   mask  = (const int*)d_in[2];
  const float* Wq    = (const float*)d_in[3];
  const float* bq    = (const float*)d_in[4];
  const float* Wk    = (const float*)d_in[5];
  const float* bk    = (const float*)d_in[6];
  const float* Wv    = (const float*)d_in[7];
  const float* bv    = (const float*)d_in[8];
  float* out = (float*)d_out;
  (void)in_sizes; (void)n_in; (void)out_size;

  const size_t need_ws = 2ull * Dn * Dn * sizeof(f16);   // 1 MB for Wk16+Wv16
  const int use16 = (d_ws != nullptr && ws_size >= need_ws) ? 1 : 0;
  f16* wk16 = (f16*)d_ws;
  f16* wv16 = wk16 + (size_t)Dn * Dn;

  if (use16) {
    convert_weights_kernel<<<dim3((Dn * Dn / 4 + 255) / 256), dim3(256), 0, stream>>>(
        Wk, Wv, wk16, wv16);
  }
  mha_fused_kernel<<<dim3(Bn), dim3(256), (unsigned)SMEM_BYTES, stream>>>(
      x_non, x_seq, mask, Wq, bq, Wk, bk, Wv, bv, wk16, wv16, use16, out);
}
